// MultiHeadLatentAttention_6021544149393
// MI455X (gfx1250) — matrix operand order repaired and checked by reading, not compiled, not run on hardware
//
#include <hip/hip_runtime.h>
#include <hip/hip_bf16.h>
#include <math.h>

// ---- problem constants (match reference) ----
#define B_    2
#define S_    2048
#define DM_   2048
#define H_    16
#define NOPE_ 128
#define RD_   64
#define VD_   128
#define QR_   1536
#define KVR_  512
#define BS_   (B_ * S_)
#define QKD_  (NOPE_ + RD_)      // 192 per-head q/k dim
#define SCALE_QK 0.07216878364870322f   // 1/sqrt(192)

typedef __attribute__((ext_vector_type(16))) __bf16 v16bf;
typedef __attribute__((ext_vector_type(8)))  __bf16 v8bf;
typedef __attribute__((ext_vector_type(8)))  float  v8f;
typedef __attribute__((ext_vector_type(4)))  unsigned int v4u;
typedef __attribute__((ext_vector_type(8)))  int v8i;
typedef __attribute__((ext_vector_type(4)))  int v4i;

// ------------------------------------------------------------------
// WMMA helpers (CDNA5 wave32, 16x16x32 bf16 -> f32)
// ------------------------------------------------------------------
static __device__ __forceinline__ v8f wmma_bf16(v16bf a, v16bf b, v8f c) {
  return __builtin_amdgcn_wmma_f32_16x16x32_bf16(false, a, false, b,
                                                 (short)0, c, false, false);
}

// A operand (16x32, row-major source): lane l holds row (l&15);
// K chunks [(l>>4)*8, +8) and [16+(l>>4)*8, +8)  (ISA 7.12.2 16-bit A layout)
static __device__ __forceinline__ v16bf load_a_frag(const __bf16* A, int ldk,
                                                    int row0, int k0, int lane) {
  int m = lane & 15, half = lane >> 4;
  const __bf16* p = A + (size_t)(row0 + m) * ldk + k0 + (half << 3);
  v8bf lo = *(const v8bf*)p;
  v8bf hi = *(const v8bf*)(p + 16);
  return __builtin_shufflevector(lo, hi, 0,1,2,3,4,5,6,7,8,9,10,11,12,13,14,15);
}

// B operand (32x16) sourced from BT[N][K] (i.e. B transposed): lane l is
// column (l&15); K chunks [(l>>4)*8, +8) and [16+(l>>4)*8, +8) (matches A).
static __device__ __forceinline__ v16bf load_bt_frag(const __bf16* BT, int ldk,
                                                     int n0, int k0, int lane) {
  int n = n0 + (lane & 15), half = lane >> 4;
  const __bf16* p = BT + (size_t)n * ldk + k0 + (half << 3);
  v8bf lo = *(const v8bf*)p;
  v8bf hi = *(const v8bf*)(p + 16);
  return __builtin_shufflevector(lo, hi, 0,1,2,3,4,5,6,7,8,9,10,11,12,13,14,15);
}

// ------------------------------------------------------------------
// TDM: one-instruction 2D tile DMA global->LDS (ISA ch.8, D# groups 0/1).
// rows x cols_k tile of bf16 (data_size=2B), row stride = stride_elems.
// Descriptor words are wave-uniform (kernel args + blockIdx).
// clang-23 toolchain: 6-arg builtin (g0 v4u, g1 v8i, g2 v4i, g3 v4i, v8i, cpol)
// ------------------------------------------------------------------
static __device__ __forceinline__ void tdm_load_2d(const __bf16* gsrc,
                                                   __bf16* ldst,
                                                   int stride_elems,
                                                   int rows, int cols_k) {
  unsigned long long ga = (unsigned long long)(uintptr_t)gsrc;
  unsigned int la = (unsigned int)(uintptr_t)ldst;  // low 32 bits = LDS addr
  v4u g0;
  g0.x = 1u;                                         // count=1, user D#
  g0.y = la;                                         // lds_addr
  g0.z = (unsigned int)ga;                           // global_addr[31:0]
  g0.w = (unsigned int)((ga >> 32) & 0x01FFFFFFu) | (2u << 30); // [56:32]|type=2
  unsigned int td0 = (unsigned int)stride_elems;     // tensor_dim0 (row length)
  unsigned int td1 = 0x00100000u;                    // tensor_dim1 (huge, no OOB)
  unsigned long long st0 = (unsigned long long)stride_elems; // dim0 stride
  v8i g1;
  g1[0] = (int)(1u << 16);                           // mask=0, data_size=1 (2B)
  g1[1] = (int)((td0 & 0xFFFFu) << 16);              // barrier=0 | dim0 lo
  g1[2] = (int)((td0 >> 16) | ((td1 & 0xFFFFu) << 16));
  g1[3] = (int)((td1 >> 16) | ((unsigned int)cols_k << 16));   // tile_dim0
  g1[4] = (int)(unsigned int)rows;                   // tile_dim1 | tile_dim2=0
  g1[5] = (int)(unsigned int)(st0 & 0xFFFFFFFFu);    // dim0_stride lo
  g1[6] = (int)(unsigned int)((st0 >> 32) & 0xFFFFu);// dim0_stride hi | s1 lo
  g1[7] = 0;                                         // dim1_stride hi
  v4i z4 = {0, 0, 0, 0};
  v8i z8 = {0, 0, 0, 0, 0, 0, 0, 0};
  __builtin_amdgcn_tensor_load_to_lds(g0, g1, z4, z4, z8, 0);
}

// ------------------------------------------------------------------
// Elementwise convert / transpose-convert
// ------------------------------------------------------------------
__global__ void k_f32_to_bf16(const float* __restrict__ in,
                              __bf16* __restrict__ out, size_t n) {
  size_t i = (size_t)blockIdx.x * blockDim.x + threadIdx.x;
  if (i < n) out[i] = (__bf16)in[i];
}

// in [K,N] f32 -> out [N,K] bf16
__global__ void k_transpose_bf16(const float* __restrict__ in,
                                 __bf16* __restrict__ out, int K, int N) {
  size_t i = (size_t)blockIdx.x * blockDim.x + threadIdx.x;
  size_t total = (size_t)K * N;
  if (i >= total) return;
  int k = (int)(i / N), n = (int)(i % N);
  out[(size_t)n * K + k] = (__bf16)in[i];
}

// ------------------------------------------------------------------
// RMSNorm: f32 rows -> bf16 rows
// ------------------------------------------------------------------
__global__ __launch_bounds__(256) void k_rmsnorm_bf16(
    const float* __restrict__ in, const float* __restrict__ w,
    __bf16* __restrict__ out, int R) {
  int row = blockIdx.x;
  const float* x = in + (size_t)row * R;
  float ss = 0.f;
  for (int i = threadIdx.x; i < R; i += 256) { float v = x[i]; ss += v * v; }
  __shared__ float red[256];
  red[threadIdx.x] = ss;
  __syncthreads();
  for (int s = 128; s > 0; s >>= 1) {
    if (threadIdx.x < (unsigned)s) red[threadIdx.x] += red[threadIdx.x + s];
    __syncthreads();
  }
  float scale = rsqrtf(red[0] / (float)R + 1e-6f);
  for (int i = threadIdx.x; i < R; i += 256)
    out[(size_t)row * R + i] = (__bf16)(x[i] * scale * w[i]);
}

// ------------------------------------------------------------------
// TDM double-buffered WMMA GEMM: C[M,N] f32 = A[M,K] * BT[N,K]^T.
// 256 threads = 8 waves (4x2), workgroup tile 128x64, K staged 64 at a time.
// Wave 0 drives the Tensor Data Mover (2 tensor_load_to_lds per stage,
// TENSORcnt double buffering); all waves consume tiles from LDS.
// Requires M%128==0, N%64==0, K%64==0.
// ------------------------------------------------------------------
__global__ __launch_bounds__(256) void k_gemm_bf16_tdm(
    const __bf16* __restrict__ A, const __bf16* __restrict__ BT,
    float* __restrict__ C, int M, int N, int K) {
  __shared__ __attribute__((aligned(16))) __bf16 tA[2][128 * 64];
  __shared__ __attribute__((aligned(16))) __bf16 tB[2][64 * 64];

  const int lane = threadIdx.x & 31;
  const int w    = threadIdx.x >> 5;
  const int row0 = blockIdx.y * 128;
  const int col0 = blockIdx.x * 64;
  const int wrow = (w >> 1) * 32;        // 0,32,64,96
  const int wcol = (w & 1) * 32;         // 0,32

  v8f acc00 = {0,0,0,0,0,0,0,0}, acc01 = acc00, acc10 = acc00, acc11 = acc00;

  const bool issuer = (threadIdx.x < 32);
  if (issuer) {
    tdm_load_2d(A  + (size_t)row0 * K, tA[0], K, 128, 64);
    tdm_load_2d(BT + (size_t)col0 * K, tB[0], K, 64, 64);
  }

  int buf = 0;
  for (int k0 = 0; k0 < K; k0 += 64) {
    if (issuer) {
      if (k0 + 64 < K) {
        tdm_load_2d(A  + (size_t)row0 * K + k0 + 64, tA[buf ^ 1], K, 128, 64);
        tdm_load_2d(BT + (size_t)col0 * K + k0 + 64, tB[buf ^ 1], K, 64, 64);
        __builtin_amdgcn_s_wait_tensorcnt((short)2);   // current bufs done
      } else {
        __builtin_amdgcn_s_wait_tensorcnt((short)0);
      }
    }
    __syncthreads();                      // tiles visible to all waves

#pragma unroll
    for (int kk = 0; kk < 64; kk += 32) {
      v16bf a0 = load_a_frag(tA[buf], 64, wrow,      kk, lane);
      v16bf a1 = load_a_frag(tA[buf], 64, wrow + 16, kk, lane);
      v16bf b0 = load_bt_frag(tB[buf], 64, wcol,      kk, lane);
      v16bf b1 = load_bt_frag(tB[buf], 64, wcol + 16, kk, lane);
      acc00 = wmma_bf16(a0, b0, acc00);
      acc01 = wmma_bf16(a0, b1, acc01);
      acc10 = wmma_bf16(a1, b0, acc10);
      acc11 = wmma_bf16(a1, b1, acc11);
    }
    __syncthreads();                      // reads done before buf reuse
    buf ^= 1;
  }

  // C/D layout: lane l, vgpr v -> row = v + 8*(l>>4), col = l&15
  int m0 = row0 + wrow + 8 * (lane >> 4);
  int n  = lane & 15;
#pragma unroll
  for (int v = 0; v < 8; ++v) {
    C[(size_t)(m0 + v)      * N + col0 + wcol + n]      = acc00[v];
    C[(size_t)(m0 + v)      * N + col0 + wcol + 16 + n] = acc01[v];
    C[(size_t)(m0 + 16 + v) * N + col0 + wcol + n]      = acc10[v];
    C[(size_t)(m0 + 16 + v) * N + col0 + wcol + 16 + n] = acc11[v];
  }
}

// ------------------------------------------------------------------
// Pack Q: [b,h,s,192] bf16, rope applied to last 64, 1/sqrt(192) folded in.
// ------------------------------------------------------------------
__global__ void k_pack_q(const float* __restrict__ qnope,
                         const float* __restrict__ qrope,
                         __bf16* __restrict__ qall) {
  size_t idx = (size_t)blockIdx.x * blockDim.x + threadIdx.x;
  size_t total = (size_t)B_ * H_ * S_ * QKD_;
  if (idx >= total) return;
  int d = (int)(idx % QKD_);
  int s = (int)((idx / QKD_) % S_);
  int h = (int)((idx / ((size_t)QKD_ * S_)) % H_);
  int b = (int)(idx / ((size_t)QKD_ * S_ * H_));
  float val;
  if (d < NOPE_) {
    val = qnope[(size_t)(b * S_ + s) * (H_ * NOPE_) + h * NOPE_ + d];
  } else {
    int dr = d - NOPE_;
    int i  = dr >> 1;
    float inv = __powf(10000.f, -(float)(2 * i) / (float)RD_);
    float ang = (float)s * inv;
    float c = __cosf(ang), sn = __sinf(ang);
    const float* base = qrope + (size_t)(b * S_ + s) * (H_ * RD_) + h * RD_;
    float x1 = base[2 * i], x2 = base[2 * i + 1];
    val = (dr & 1) ? (x1 * sn + x2 * c) : (x1 * c - x2 * sn);
  }
  qall[idx] = (__bf16)(val * SCALE_QK);
}

// Pack K: [b,h,s,192] bf16 (nope from kv_up, rope part shared across heads)
__global__ void k_pack_k(const float* __restrict__ kvup,
                         const float* __restrict__ krope,
                         __bf16* __restrict__ kall) {
  size_t idx = (size_t)blockIdx.x * blockDim.x + threadIdx.x;
  size_t total = (size_t)B_ * H_ * S_ * QKD_;
  if (idx >= total) return;
  int d = (int)(idx % QKD_);
  int s = (int)((idx / QKD_) % S_);
  int h = (int)((idx / ((size_t)QKD_ * S_)) % H_);
  int b = (int)(idx / ((size_t)QKD_ * S_ * H_));
  float val;
  if (d < NOPE_) {
    val = kvup[(size_t)(b * S_ + s) * (H_ * (NOPE_ + VD_)) + h * (NOPE_ + VD_) + d];
  } else {
    int dr = d - NOPE_;
    int i  = dr >> 1;
    float inv = __powf(10000.f, -(float)(2 * i) / (float)RD_);
    float ang = (float)s * inv;
    float c = __cosf(ang), sn = __sinf(ang);
    const float* base = krope + (size_t)(b * S_ + s) * RD_;
    float x1 = base[2 * i], x2 = base[2 * i + 1];
    val = (dr & 1) ? (x1 * sn + x2 * c) : (x1 * c - x2 * sn);
  }
  kall[idx] = (__bf16)val;
}

// Pack V transposed: vT[b,h,d,s] bf16  (so PV B-operand loads are contiguous)
__global__ void k_pack_vt(const float* __restrict__ kvup,
                          __bf16* __restrict__ vT) {
  size_t idx = (size_t)blockIdx.x * blockDim.x + threadIdx.x;
  size_t total = (size_t)B_ * H_ * VD_ * S_;
  if (idx >= total) return;
  int s = (int)(idx % S_);
  int d = (int)((idx / S_) % VD_);
  int h = (int)((idx / ((size_t)S_ * VD_)) % H_);
  int b = (int)(idx / ((size_t)S_ * VD_ * H_));
  vT[idx] = (__bf16)kvup[(size_t)(b * S_ + s) * (H_ * (NOPE_ + VD_)) +
                         h * (NOPE_ + VD_) + NOPE_ + d];
}

// ------------------------------------------------------------------
// Flash attention: one wave per (b, h, 16-row q-tile).
// Per 32-key block: 12 WMMA for scores, online softmax (half-wave shfl
// butterflies), P via LDS transpose, 8 WMMA for P*V.
// ------------------------------------------------------------------
__global__ __launch_bounds__(32) void k_mla_attn(
    const __bf16* __restrict__ qall, const __bf16* __restrict__ kall,
    const __bf16* __restrict__ vT,   __bf16* __restrict__ attn) {
  const int lane = threadIdx.x;
  const int half = lane >> 4;
  const int nlo  = lane & 15;
  int tile = blockIdx.x;
  int qt = tile % (S_ / 16);
  int h  = (tile / (S_ / 16)) % H_;
  int b  = tile / ((S_ / 16) * H_);
  int q0 = qt * 16;

  const __bf16* Q  = qall + ((size_t)(b * H_ + h) * S_) * QKD_;
  const __bf16* Kp = kall + ((size_t)(b * H_ + h) * S_) * QKD_;
  const __bf16* V  = vT   + ((size_t)(b * H_ + h) * VD_) * S_;

  v16bf qf[6];
#pragma unroll
  for (int c = 0; c < 6; ++c) qf[c] = load_a_frag(Q, QKD_, q0, c * 32, lane);

  v8f o[8];
#pragma unroll
  for (int j = 0; j < 8; ++j) o[j] = (v8f){0,0,0,0,0,0,0,0};
  float mrow[8], lrow[8];
#pragma unroll
  for (int v = 0; v < 8; ++v) { mrow[v] = -1e30f; lrow[v] = 0.f; }

  __shared__ __attribute__((aligned(16))) __bf16 p_lds[16 * 32];

  const int nEnd = q0 + 16;              // causal: keys 0 .. q0+15
  for (int kt = 0; kt < nEnd; kt += 32) {
    v8f s0 = (v8f){0,0,0,0,0,0,0,0};
    v8f s1 = s0;
#pragma unroll
    for (int c = 0; c < 6; ++c) {
      v16bf kb0 = load_bt_frag(Kp, QKD_, kt,      c * 32, lane);
      v16bf kb1 = load_bt_frag(Kp, QKD_, kt + 16, c * 32, lane);
      s0 = wmma_bf16(qf[c], kb0, s0);
      s1 = wmma_bf16(qf[c], kb1, s1);
    }
    int ki0 = kt + nlo, ki1 = ki0 + 16;
    float p0[8], p1[8], corr[8];
#pragma unroll
    for (int v = 0; v < 8; ++v) {
      int qi = q0 + v + 8 * half;
      float a  = (ki0 <= qi) ? s0[v] : -1e30f;
      float bb = (ki1 <= qi) ? s1[v] : -1e30f;
      s0[v] = a; s1[v] = bb;
      float rm = fmaxf(a, bb);
#pragma unroll
      for (int msk = 1; msk < 16; msk <<= 1)
        rm = fmaxf(rm, __shfl_xor(rm, msk, 32));
      float mnew = fmaxf(mrow[v], rm);
      p0[v] = __expf(s0[v] - mnew);
      p1[v] = __expf(s1[v] - mnew);
      float rs = p0[v] + p1[v];
#pragma unroll
      for (int msk = 1; msk < 16; msk <<= 1)
        rs += __shfl_xor(rs, msk, 32);
      corr[v] = __expf(mrow[v] - mnew);
      lrow[v] = lrow[v] * corr[v] + rs;
      mrow[v] = mnew;
    }
#pragma unroll
    for (int j = 0; j < 8; ++j)
#pragma unroll
      for (int v = 0; v < 8; ++v) o[j][v] *= corr[v];

#pragma unroll
    for (int v = 0; v < 8; ++v) {
      int m = v + 8 * half;
      p_lds[m * 32 + nlo]      = (__bf16)p0[v];
      p_lds[m * 32 + 16 + nlo] = (__bf16)p1[v];
    }
    __syncthreads();
    v16bf pf;
    {
      const __bf16* pp = p_lds + nlo * 32 + (half << 3);
      v8bf lo = *(const v8bf*)pp;
      v8bf hi = *(const v8bf*)(pp + 16);
      pf = __builtin_shufflevector(lo, hi, 0,1,2,3,4,5,6,7,8,9,10,11,12,13,14,15);
    }
    __syncthreads();

#pragma unroll
    for (int j = 0; j < 8; ++j) {
      v16bf vb = load_bt_frag(V, S_, j * 16, kt, lane);
      o[j] = wmma_bf16(pf, vb, o[j]);
    }
  }

#pragma unroll
  for (int v = 0; v < 8; ++v) {
    float invl = 1.0f / lrow[v];
    int srow = q0 + v + 8 * half;
    __bf16* dst = attn + (size_t)(b * S_ + srow) * (H_ * VD_) + h * VD_;
#pragma unroll
    for (int j = 0; j < 8; ++j) dst[j * 16 + nlo] = (__bf16)(o[j][v] * invl);
  }
}

// ------------------------------------------------------------------
// Host-side orchestration
// ------------------------------------------------------------------
static inline char* wsalloc(char*& p, size_t bytes) {
  char* r = p;
  p += (bytes + 255) & ~(size_t)255;
  return r;
}

extern "C" void kernel_launch(void* const* d_in, const int* in_sizes, int n_in,
                              void* d_out, int out_size, void* d_ws, size_t ws_size,
                              hipStream_t stream) {
  (void)in_sizes; (void)n_in; (void)out_size; (void)ws_size;
  const float* x        = (const float*)d_in[0];
  const float* wq_down  = (const float*)d_in[1];
  const float* q_norm_w = (const float*)d_in[2];
  const float* wq_up    = (const float*)d_in[3];
  const float* wq_rope  = (const float*)d_in[4];
  const float* wkv_down = (const float*)d_in[5];
  const float* kv_norm_w= (const float*)d_in[6];
  const float* wkv_up   = (const float*)d_in[7];
  const float* wk_rope  = (const float*)d_in[8];
  const float* wo       = (const float*)d_in[9];
  float* out = (float*)d_out;

  char* p = (char*)d_ws;
  __bf16* x_bf    = (__bf16*)wsalloc(p, (size_t)BS_ * DM_ * 2);
  __bf16* wqd_t   = (__bf16*)wsalloc(p, (size_t)QR_ * DM_ * 2);
  __bf16* wqu_t   = (__bf16*)wsalloc(p, (size_t)(H_ * NOPE_) * QR_ * 2);
  __bf16* wqr_t   = (__bf16*)wsalloc(p, (size_t)(H_ * RD_) * QR_ * 2);
  __bf16* wkvd_t  = (__bf16*)wsalloc(p, (size_t)KVR_ * DM_ * 2);
  __bf16* wkvu_t  = (__bf16*)wsalloc(p, (size_t)(H_ * (NOPE_ + VD_)) * KVR_ * 2);
  __bf16* wkr_t   = (__bf16*)wsalloc(p, (size_t)RD_ * DM_ * 2);
  __bf16* wo_t    = (__bf16*)wsalloc(p, (size_t)DM_ * (H_ * VD_) * 2);
  float*  qpre    = (float*)wsalloc(p, (size_t)BS_ * QR_ * 4);
  float*  kvpre   = (float*)wsalloc(p, (size_t)BS_ * KVR_ * 4);
  __bf16* qc_bf   = (__bf16*)wsalloc(p, (size_t)BS_ * QR_ * 2);
  __bf16* kvc_bf  = (__bf16*)wsalloc(p, (size_t)BS_ * KVR_ * 2);
  float*  qnope   = (float*)wsalloc(p, (size_t)BS_ * H_ * NOPE_ * 4);
  float*  qrope   = (float*)wsalloc(p, (size_t)BS_ * H_ * RD_ * 4);
  float*  kvup    = (float*)wsalloc(p, (size_t)BS_ * H_ * (NOPE_ + VD_) * 4);
  float*  krope   = (float*)wsalloc(p, (size_t)BS_ * RD_ * 4);
  __bf16* qall    = (__bf16*)wsalloc(p, (size_t)B_ * H_ * S_ * QKD_ * 2);
  __bf16* kall    = (__bf16*)wsalloc(p, (size_t)B_ * H_ * S_ * QKD_ * 2);
  __bf16* vT      = (__bf16*)wsalloc(p, (size_t)B_ * H_ * VD_ * S_ * 2);
  __bf16* attn_bf = (__bf16*)wsalloc(p, (size_t)BS_ * H_ * VD_ * 2);

  auto blocks = [](size_t n) { return (unsigned)((n + 255) / 256); };

  // 1) convert x; transpose-convert all weights to bf16 [N,K]
  {
    size_t n = (size_t)BS_ * DM_;
    k_f32_to_bf16<<<blocks(n), 256, 0, stream>>>(x, x_bf, n);
  }
  k_transpose_bf16<<<blocks((size_t)DM_ * QR_), 256, 0, stream>>>(wq_down, wqd_t, DM_, QR_);
  k_transpose_bf16<<<blocks((size_t)QR_ * H_ * NOPE_), 256, 0, stream>>>(wq_up, wqu_t, QR_, H_ * NOPE_);
  k_transpose_bf16<<<blocks((size_t)QR_ * H_ * RD_), 256, 0, stream>>>(wq_rope, wqr_t, QR_, H_ * RD_);
  k_transpose_bf16<<<blocks((size_t)DM_ * KVR_), 256, 0, stream>>>(wkv_down, wkvd_t, DM_, KVR_);
  k_transpose_bf16<<<blocks((size_t)KVR_ * H_ * (NOPE_ + VD_)), 256, 0, stream>>>(wkv_up, wkvu_t, KVR_, H_ * (NOPE_ + VD_));
  k_transpose_bf16<<<blocks((size_t)DM_ * RD_), 256, 0, stream>>>(wk_rope, wkr_t, DM_, RD_);
  k_transpose_bf16<<<blocks((size_t)DM_ * DM_), 256, 0, stream>>>(wo, wo_t, H_ * VD_, DM_);

  // 2) down projections + k_rope raw (TDM double-buffered GEMMs)
  dim3 blk(256);
  k_gemm_bf16_tdm<<<dim3(QR_ / 64, BS_ / 128), blk, 0, stream>>>(x_bf, wqd_t, qpre, BS_, QR_, DM_);
  k_gemm_bf16_tdm<<<dim3(KVR_ / 64, BS_ / 128), blk, 0, stream>>>(x_bf, wkvd_t, kvpre, BS_, KVR_, DM_);
  k_gemm_bf16_tdm<<<dim3(1, BS_ / 128), blk, 0, stream>>>(x_bf, wkr_t, krope, BS_, RD_, DM_);

  // 3) rmsnorms -> bf16
  k_rmsnorm_bf16<<<BS_, 256, 0, stream>>>(qpre, q_norm_w, qc_bf, QR_);
  k_rmsnorm_bf16<<<BS_, 256, 0, stream>>>(kvpre, kv_norm_w, kvc_bf, KVR_);

  // 4) up projections
  k_gemm_bf16_tdm<<<dim3((H_ * NOPE_) / 64, BS_ / 128), blk, 0, stream>>>(qc_bf, wqu_t, qnope, BS_, H_ * NOPE_, QR_);
  k_gemm_bf16_tdm<<<dim3((H_ * RD_) / 64, BS_ / 128), blk, 0, stream>>>(qc_bf, wqr_t, qrope, BS_, H_ * RD_, QR_);
  k_gemm_bf16_tdm<<<dim3((H_ * (NOPE_ + VD_)) / 64, BS_ / 128), blk, 0, stream>>>(kvc_bf, wkvu_t, kvup, BS_, H_ * (NOPE_ + VD_), KVR_);

  // 5) pack Q/K (rope) and V^T as bf16
  {
    size_t n = (size_t)B_ * H_ * S_ * QKD_;
    k_pack_q<<<blocks(n), 256, 0, stream>>>(qnope, qrope, qall);
    k_pack_k<<<blocks(n), 256, 0, stream>>>(kvup, krope, kall);
    size_t nv = (size_t)B_ * H_ * VD_ * S_;
    k_pack_vt<<<blocks(nv), 256, 0, stream>>>(kvup, vT);
  }

  // 6) flash attention (one wave per 16-row q tile)
  k_mla_attn<<<B_ * H_ * (S_ / 16), 32, 0, stream>>>(qall, kall, vT, attn_bf);

  // 7) output projection -> d_out (f32)
  k_gemm_bf16_tdm<<<dim3(DM_ / 64, BS_ / 128), blk, 0, stream>>>(attn_bf, wo_t, out, BS_, DM_, H_ * VD_);
}
